// CausalSelfAttention_90245852823704
// MI455X (gfx1250) — matrix operand order repaired and checked by reading, not compiled, not run on hardware
//
#include <hip/hip_runtime.h>

// ---------------------------------------------------------------------------
// Problem constants (match reference)
// ---------------------------------------------------------------------------
#define B_   2
#define S_   2048
#define D_   2048
#define H_   16
#define KV_  4
#define DK_  128          // head dim
#define KVD_ 1024         // 2*KV*dk
#define BS_  (B_ * S_)    // 4096

typedef __bf16 bf16;
typedef __attribute__((ext_vector_type(16))) __bf16 v16bf;
typedef __attribute__((ext_vector_type(8)))  __bf16 v8bf;
typedef __attribute__((ext_vector_type(8)))  float  v8f;

__device__ __forceinline__ v8f vzero8f() {
    v8f z = {0.f, 0.f, 0.f, 0.f, 0.f, 0.f, 0.f, 0.f};
    return z;
}

__device__ __forceinline__ v8f wmma_bf16(v16bf a, v16bf b, v8f c) {
    return __builtin_amdgcn_wmma_f32_16x16x32_bf16(
        /*neg_a=*/false, a, /*neg_b=*/false, b,
        /*c_mod=*/(short)0, c, /*reuse_a=*/false, /*reuse_b=*/false);
}

// A-matrix 16x32 bf16 fragment (row-major source, leading dim ld elements).
// Lane l holds row m=l%16; lanes<16: K 0..15, lanes>=16: K 16..31 (matches load_b_frag).
__device__ __forceinline__ v16bf load_a_frag(const bf16* base, int ld, int lane) {
    int m  = lane & 15;
    int kb = (lane >> 4) << 4;
    const bf16* p = base + (size_t)m * ld + kb;
    v8bf lo = *(const v8bf*)(p);
    v8bf hi = *(const v8bf*)(p + 8);
    return __builtin_shufflevector(lo, hi, 0,1,2,3,4,5,6,7,8,9,10,11,12,13,14,15);
}

// B-matrix 32x16 bf16 fragment. Source is "column-of-B contiguous along K":
// element (k,n) = src[n*ld + k]. Lane l holds column n=l%16, K klo..klo+15 contiguous.
__device__ __forceinline__ v16bf load_b_frag(const bf16* base, int ld, int lane) {
    int n   = lane & 15;
    int klo = (lane >> 4) << 4;
    return *(const v16bf*)(base + (size_t)n * ld + klo);
}

// ---------------------------------------------------------------------------
// fp32 -> bf16 conversion
// ---------------------------------------------------------------------------
__global__ void cvt_f32_bf16(const float* __restrict__ in, bf16* __restrict__ out, int n) {
    int i = blockIdx.x * blockDim.x + threadIdx.x;
    if (i < n) out[i] = (bf16)in[i];
}

// ---------------------------------------------------------------------------
// GEMM: C[M,N] = A[M,K] * W[N,K]^T   (A,W bf16 row-major, C fp32 row-major)
// One wave computes a 32x32 tile. Grid supplies exactly (M/32)*(N/32) waves.
// ---------------------------------------------------------------------------
__global__ void __launch_bounds__(256)
gemm_bf16_nt(const bf16* __restrict__ A, const bf16* __restrict__ W,
             float* __restrict__ C, int M, int N, int K) {
    int lane   = threadIdx.x & 31;
    int w      = (blockIdx.x * blockDim.x + threadIdx.x) >> 5;
    int tilesN = N >> 5;
    int tm = (w / tilesN) << 5;
    int tn = (w % tilesN) << 5;

    const bf16* a0p = A + (size_t)tm * K;
    const bf16* a1p = A + (size_t)(tm + 16) * K;
    const bf16* b0p = W + (size_t)tn * K;
    const bf16* b1p = W + (size_t)(tn + 16) * K;

    v8f c00 = vzero8f(), c01 = vzero8f(), c10 = vzero8f(), c11 = vzero8f();

    for (int k = 0; k < K; k += 32) {
        __builtin_prefetch(a0p + k + 256, 0, 1);
        __builtin_prefetch(b0p + k + 256, 0, 1);
        v16bf a0 = load_a_frag(a0p + k, K, lane);
        v16bf a1 = load_a_frag(a1p + k, K, lane);
        v16bf b0 = load_b_frag(b0p + k, K, lane);
        v16bf b1 = load_b_frag(b1p + k, K, lane);
        c00 = wmma_bf16(a0, b0, c00);
        c01 = wmma_bf16(a0, b1, c01);
        c10 = wmma_bf16(a1, b0, c10);
        c11 = wmma_bf16(a1, b1, c11);
    }

    int mb = (lane >> 4) << 3;
    int n  = lane & 15;
    float* crow = C + (size_t)(tm + mb) * N + tn;
#pragma unroll
    for (int i = 0; i < 8; ++i) {
        crow[(size_t)i * N + n]             = c00[i];
        crow[(size_t)i * N + 16 + n]        = c01[i];
        crow[(size_t)(16 + i) * N + n]      = c10[i];
        crow[(size_t)(16 + i) * N + 16 + n] = c11[i];
    }
}

// ---------------------------------------------------------------------------
// RoPE on Q: qlin fp32 [B*S, D] -> qb bf16 [B,H,S,dk]
// ---------------------------------------------------------------------------
__global__ void rope_q_kernel(const float* __restrict__ qlin, bf16* __restrict__ qb) {
    int i = blockIdx.x * blockDim.x + threadIdx.x;       // B*S*H*64 threads
    if (i >= B_ * S_ * H_ * (DK_ / 2)) return;
    int p = i & 63;
    int h = (i >> 6) & 15;
    int s = (i >> 10) & 2047;
    int b = i >> 21;

    float inv_freq = powf(10000.0f, -(float)(2 * p) / (float)DK_);
    float ang = (float)s * inv_freq;
    float sn, cs;
    sincosf(ang, &sn, &cs);

    size_t src = ((size_t)(b * S_ + s)) * D_ + h * DK_ + 2 * p;
    float e = qlin[src], o = qlin[src + 1];
    size_t dst = (((size_t)(b * H_ + h)) * S_ + s) * DK_ + 2 * p;
    qb[dst]     = (bf16)(e * cs - o * sn);
    qb[dst + 1] = (bf16)(e * sn + o * cs);
}

// ---------------------------------------------------------------------------
// RoPE on K + transpose V: kvlin fp32 [B*S, 2*KV*dk]
//   -> kb bf16 [B,KV,S,dk] (roped),  vt bf16 [B,KV,dk,S] (transposed)
// ---------------------------------------------------------------------------
__global__ void rope_kv_kernel(const float* __restrict__ kvlin,
                               bf16* __restrict__ kb, bf16* __restrict__ vt) {
    int i = blockIdx.x * blockDim.x + threadIdx.x;       // B*S*KV*64 threads
    if (i >= B_ * S_ * KV_ * (DK_ / 2)) return;
    int p = i & 63;
    int h = (i >> 6) & 3;
    int s = (i >> 8) & 2047;
    int b = i >> 19;

    float inv_freq = powf(10000.0f, -(float)(2 * p) / (float)DK_);
    float ang = (float)s * inv_freq;
    float sn, cs;
    sincosf(ang, &sn, &cs);

    size_t row  = ((size_t)(b * S_ + s)) * KVD_;
    size_t ksrc = row + h * DK_ + 2 * p;
    float e = kvlin[ksrc], o = kvlin[ksrc + 1];
    size_t kdst = (((size_t)(b * KV_ + h)) * S_ + s) * DK_ + 2 * p;
    kb[kdst]     = (bf16)(e * cs - o * sn);
    kb[kdst + 1] = (bf16)(e * sn + o * cs);

    size_t vsrc = row + KV_ * DK_ + h * DK_ + 2 * p;
    float ve = kvlin[vsrc], vo = kvlin[vsrc + 1];
    size_t vbase = ((size_t)(b * KV_ + h)) * DK_;
    vt[(vbase + 2 * p) * (size_t)S_ + s]     = (bf16)ve;
    vt[(vbase + 2 * p + 1) * (size_t)S_ + s] = (bf16)vo;
}

// ---------------------------------------------------------------------------
// Flash attention: one wave per (b, h, 16-query tile).
// Q bf16 [B,H,S,dk], K bf16 [B,KV,S,dk], Vt bf16 [B,KV,dk,S]
// -> Ob bf16 [B*S, D] (head-major columns, ready for output projection).
// ---------------------------------------------------------------------------
__global__ void __launch_bounds__(128)
flash_attn_kernel(const bf16* __restrict__ Qb, const bf16* __restrict__ Kb,
                  const bf16* __restrict__ Vt, bf16* __restrict__ Ob) {
    __shared__ bf16 plds[4][16 * 32];

    int lane = threadIdx.x & 31;
    int wib  = threadIdx.x >> 5;
    int w    = blockIdx.x * 4 + wib;

    int qt = w & 127;             // S/16 tiles
    int h  = (w >> 7) & 15;
    int b  = w >> 11;
    int hk = h >> 2;              // GQA: 4 query heads per kv head
    int qbase = qt * 16;

    const bf16* qptr  = Qb + (((size_t)(b * H_ + h)) * S_ + qbase) * DK_;
    const bf16* kbase = Kb + ((size_t)(b * KV_ + hk)) * S_ * DK_;
    const bf16* vbase = Vt + ((size_t)(b * KV_ + hk)) * DK_ * S_;

    v16bf qa[4];
#pragma unroll
    for (int t = 0; t < 4; ++t) qa[t] = load_a_frag(qptr + t * 32, DK_, lane);

    int mb = (lane >> 4) << 3;
    int n  = lane & 15;

    float mrow[8], lrow[8];
    v8f o[8];
#pragma unroll
    for (int i = 0; i < 8; ++i) { mrow[i] = -1e30f; lrow[i] = 0.f; o[i] = vzero8f(); }

    const float scale = 0.08838834764831845f;   // 1/sqrt(128)
    bf16* pl = plds[wib];

    for (int j0 = 0; j0 < qbase + 16; j0 += 32) {
        // ---- logits: S = Q (16x128) . K^T, two 16-key column tiles ----
        v8f s0 = vzero8f(), s1 = vzero8f();
#pragma unroll
        for (int t = 0; t < 4; ++t) {
            v16bf bk0 = load_b_frag(kbase + (size_t)j0 * DK_ + t * 32, DK_, lane);
            v16bf bk1 = load_b_frag(kbase + (size_t)(j0 + 16) * DK_ + t * 32, DK_, lane);
            s0 = wmma_bf16(qa[t], bk0, s0);
            s1 = wmma_bf16(qa[t], bk1, s1);
        }

        // ---- scale + causal mask + row max ----
        float mx[8];
#pragma unroll
        for (int i = 0; i < 8; ++i) {
            int qrow = qbase + mb + i;
            float a0 = s0[i] * scale;
            float a1 = s1[i] * scale;
            if (j0 + n > qrow)      a0 = -1e30f;
            if (j0 + 16 + n > qrow) a1 = -1e30f;
            s0[i] = a0; s1[i] = a1;
            mx[i] = fmaxf(a0, a1);
        }
#pragma unroll
        for (int i = 0; i < 8; ++i) {
#pragma unroll
            for (int d = 1; d < 16; d <<= 1)
                mx[i] = fmaxf(mx[i], __shfl_xor(mx[i], d, 32));
        }

        // ---- online softmax update ----
        float alpha[8], rs[8];
#pragma unroll
        for (int i = 0; i < 8; ++i) {
            float mn = fmaxf(mrow[i], mx[i]);
            alpha[i] = __expf(mrow[i] - mn);
            mrow[i]  = mn;
            float p0 = __expf(s0[i] - mn);
            float p1 = __expf(s1[i] - mn);
            s0[i] = p0; s1[i] = p1;
            rs[i] = p0 + p1;
        }
#pragma unroll
        for (int i = 0; i < 8; ++i) {
#pragma unroll
            for (int d = 1; d < 16; d <<= 1)
                rs[i] += __shfl_xor(rs[i], d, 32);
        }
#pragma unroll
        for (int i = 0; i < 8; ++i) lrow[i] = lrow[i] * alpha[i] + rs[i];
#pragma unroll
        for (int t = 0; t < 8; ++t)
#pragma unroll
            for (int i = 0; i < 8; ++i) o[t][i] *= alpha[i];

        // ---- transpose P (C-layout) -> A-fragment layout through LDS ----
#pragma unroll
        for (int i = 0; i < 8; ++i) {
            pl[(mb + i) * 32 + n]      = (bf16)s0[i];
            pl[(mb + i) * 32 + 16 + n] = (bf16)s1[i];
        }
        asm volatile("s_wait_dscnt 0" ::: "memory");
        v16bf pa = load_a_frag(pl, 32, lane);
        asm volatile("s_wait_dscnt 0" ::: "memory");

        // ---- O += P (16x32) . V (32x128), 8 dk-column tiles ----
#pragma unroll
        for (int t = 0; t < 8; ++t) {
            v16bf bv = load_b_frag(vbase + (size_t)(t * 16) * S_ + j0, S_, lane);
            o[t] = wmma_bf16(pa, bv, o[t]);
        }
    }

    // ---- normalize and write [B*S, D] with column offset h*dk ----
    bf16* orow = Ob + ((size_t)(b * S_ + qbase)) * D_ + h * DK_;
#pragma unroll
    for (int t = 0; t < 8; ++t)
#pragma unroll
        for (int i = 0; i < 8; ++i) {
            float v = o[t][i] / lrow[i];
            orow[(size_t)(mb + i) * D_ + t * 16 + n] = (bf16)v;
        }
}

// ---------------------------------------------------------------------------
// Launcher
// ---------------------------------------------------------------------------
extern "C" void kernel_launch(void* const* d_in, const int* in_sizes, int n_in,
                              void* d_out, int out_size, void* d_ws, size_t ws_size,
                              hipStream_t stream) {
    const float* x   = (const float*)d_in[0];
    const float* Wq  = (const float*)d_in[1];
    const float* Wkv = (const float*)d_in[2];
    const float* Wc  = (const float*)d_in[3];
    float* out = (float*)d_out;

    char* ws = (char*)d_ws;
    bf16* xb    = (bf16*)ws;                 ws += (size_t)BS_ * D_ * 2;       // 16 MB
    bf16* wqb   = (bf16*)ws;                 ws += (size_t)D_ * D_ * 2;        // 8 MB
    bf16* wkvb  = (bf16*)ws;                 ws += (size_t)KVD_ * D_ * 2;      // 4 MB
    bf16* wcb   = (bf16*)ws;                 ws += (size_t)D_ * D_ * 2;        // 8 MB
    float* qlin = (float*)ws;                ws += (size_t)BS_ * D_ * 4;       // 32 MB
    float* kvlin= (float*)ws;                ws += (size_t)BS_ * KVD_ * 4;     // 16 MB
    bf16* qb    = (bf16*)ws;                 ws += (size_t)BS_ * D_ * 2;       // 16 MB
    bf16* kb    = (bf16*)ws;                 ws += (size_t)B_ * KV_ * S_ * DK_ * 2; // 4 MB
    bf16* vt    = (bf16*)ws;                 ws += (size_t)B_ * KV_ * DK_ * S_ * 2; // 4 MB
    bf16* attnb = (bf16*)ws;                 ws += (size_t)BS_ * D_ * 2;       // 16 MB

    // 1) fp32 -> bf16 conversions
    {
        int n;
        n = BS_ * D_;     cvt_f32_bf16<<<(n + 255) / 256, 256, 0, stream>>>(x,   xb,   n);
        n = D_ * D_;      cvt_f32_bf16<<<(n + 255) / 256, 256, 0, stream>>>(Wq,  wqb,  n);
        n = KVD_ * D_;    cvt_f32_bf16<<<(n + 255) / 256, 256, 0, stream>>>(Wkv, wkvb, n);
        n = D_ * D_;      cvt_f32_bf16<<<(n + 255) / 256, 256, 0, stream>>>(Wc,  wcb,  n);
    }

    // 2) Q projection: qlin[BS, D] = xb . Wq^T  -> 8192 waves
    gemm_bf16_nt<<<(BS_ / 32) * (D_ / 32) * 32 / 256, 256, 0, stream>>>(
        xb, wqb, qlin, BS_, D_, D_);

    // 3) KV projection: kvlin[BS, 1024] = xb . Wkv^T -> 4096 waves
    gemm_bf16_nt<<<(BS_ / 32) * (KVD_ / 32) * 32 / 256, 256, 0, stream>>>(
        xb, wkvb, kvlin, BS_, KVD_, D_);

    // 4) RoPE Q, RoPE K + transpose V
    {
        int nq = B_ * S_ * H_ * (DK_ / 2);
        rope_q_kernel<<<(nq + 255) / 256, 256, 0, stream>>>(qlin, qb);
        int nk = B_ * S_ * KV_ * (DK_ / 2);
        rope_kv_kernel<<<(nk + 255) / 256, 256, 0, stream>>>(kvlin, kb, vt);
    }

    // 5) Flash attention: 4096 waves (one per b,h,16-query tile), 4 waves/block
    flash_attn_kernel<<<(B_ * H_ * (S_ / 16)) / 4, 128, 0, stream>>>(qb, kb, vt, attnb);

    // 6) Output projection: out[BS, D] = attnb . Wc^T (fp32 out)
    gemm_bf16_nt<<<(BS_ / 32) * (D_ / 32) * 32 / 256, 256, 0, stream>>>(
        attnb, wcb, out, BS_, D_, D_);
}
